// MultiHeadDotProductSelfAttention_23708219474092
// MI455X (gfx1250) — compile-verified, matrix operand order checked by reading, not run on hardware
//
#include <hip/hip_runtime.h>

// ---------------------------------------------------------------------------
// Fused multi-head self-attention for MI455X (gfx1250, wave32).
//   B=8, N=2048, D=512, H=8, dh=64.
// cvt -> QKV GEMM (bf16 WMMA, fully-unrolled pipelined K loop) ->
// flash-attention (async-to-LDS K/V staging, bf16 WMMA, fp32 online softmax)
// -> out-proj GEMM -> fp32 out.
// ---------------------------------------------------------------------------

#define CDNA5_USE_ASYNC_LDS 1

typedef __bf16 bf16_t;
typedef __attribute__((ext_vector_type(16))) __bf16 v16bf;
typedef __attribute__((ext_vector_type(8)))  __bf16 v8bf;
typedef __attribute__((ext_vector_type(8)))  float  v8f;

#define WMMA_BF16(a, bm, c) \
  __builtin_amdgcn_wmma_f32_16x16x32_bf16(false, (a), false, (bm), (short)0, (c), false, false)

// Load a 16x32 bf16 tile in the CDNA5 A/B VGPR layout:
// lane&15 selects the row (A) / column (B); lane>>4 selects the contraction
// half: VGPR0-3 hold K = half*8 .. half*8+7, VGPR4-7 hold K = 16+half*8 ...
static __device__ __forceinline__ v16bf load_tile(const bf16_t* base, int stride, int lane) {
  const bf16_t* p = base + (size_t)(lane & 15) * stride + ((lane >> 4) << 3);
  v8bf lo = *(const v8bf*)p;
  v8bf hi = *(const v8bf*)(p + 16);
  v16bf r;
#pragma unroll
  for (int i = 0; i < 8; ++i) { r[i] = lo[i]; r[i + 8] = hi[i]; }
  return r;
}

// 16-byte async DMA global -> LDS (GLOBAL_LOAD_ASYNC_TO_LDS_B128, ASYNCcnt).
// LDS address = low 32 bits of the generic pointer (ISA 10.2).
static __device__ __forceinline__ void async_cp16(void* lds_ptr, const void* gptr) {
#if CDNA5_USE_ASYNC_LDS
  unsigned l = (unsigned)(size_t)lds_ptr;
  unsigned long long g = (unsigned long long)(size_t)gptr;
  asm volatile("global_load_async_to_lds_b128 %0, %1, off"
               :: "v"(l), "v"(g) : "memory");
#else
  *(v8bf*)lds_ptr = *(const v8bf*)gptr;
#endif
}

#if CDNA5_USE_ASYNC_LDS
#define WAIT_ASYNC_LE4() asm volatile("s_wait_asynccnt 0x4" ::: "memory")
#define WAIT_ASYNC_0()   asm volatile("s_wait_asynccnt 0x0" ::: "memory")
#else
#define WAIT_ASYNC_LE4()
#define WAIT_ASYNC_0()
#endif

// ---------------------------------------------------------------------------
// f32 -> bf16 conversion kernels
// ---------------------------------------------------------------------------
__global__ void cvt_kernel(const float* __restrict__ src, bf16_t* __restrict__ dst, int n) {
  int i = blockIdx.x * blockDim.x + threadIdx.x;
  if (i < n) dst[i] = (bf16_t)src[i];
}

__global__ void cvtT_kernel(const float* __restrict__ src, bf16_t* __restrict__ dst,
                            int rows, int cols) {
  int i = blockIdx.x * blockDim.x + threadIdx.x;
  if (i < rows * cols) {
    int r = i / cols, c = i - r * cols;
    dst[(size_t)c * rows + r] = (bf16_t)src[i];
  }
}

// ---------------------------------------------------------------------------
// QKV projection: xb[16384,512] @ Wqkv_t[1536,512]^T + bias, scattered into
// Q[b,h,n,64], K[b,h,n,64], Vt[b,h,64,n] (bf16).
// 32x64 tile per wave; K loop fully unrolled (16 steps, 8 WMMA each) so the
// scheduler pipelines loads across steps with no buffer-rotation copies.
// ---------------------------------------------------------------------------
__global__ __launch_bounds__(128) void qkv_gemm_kernel(const bf16_t* __restrict__ xb,
                                                       const bf16_t* __restrict__ Wt,
                                                       const float* __restrict__ bias,
                                                       bf16_t* __restrict__ Qo,
                                                       bf16_t* __restrict__ Ko,
                                                       bf16_t* __restrict__ Vto) {
  const int lane = threadIdx.x & 31;
  const int wave = blockIdx.x * 4 + (threadIdx.x >> 5);
  const int mt = wave / 24;              // 512 row tiles of 32
  const int nt = wave - mt * 24;         // 24 col tiles of 64
  const int hv = lane >> 4, ln = lane & 15;
  const bf16_t* arow = xb + (size_t)mt * 32 * 512;
  const bf16_t* wrow = Wt + (size_t)nt * 64 * 512;

  v8f acc[2][4] = {};
#pragma unroll
  for (int k0 = 0; k0 < 512; k0 += 32) {
    const v16bf a0t = load_tile(arow + k0, 512, lane);
    const v16bf a1t = load_tile(arow + (size_t)16 * 512 + k0, 512, lane);
#pragma unroll
    for (int g = 0; g < 4; ++g) {
      const v16bf w = load_tile(wrow + (size_t)g * 16 * 512 + k0, 512, lane);
      acc[0][g] = WMMA_BF16(a0t, w, acc[0][g]);
      acc[1][g] = WMMA_BF16(a1t, w, acc[1][g]);
    }
  }

#pragma unroll
  for (int r = 0; r < 2; ++r) {
#pragma unroll
    for (int g = 0; g < 4; ++g) {
#pragma unroll
      for (int i = 0; i < 8; ++i) {
        const int col  = nt * 64 + g * 16 + ln;
        const int grow = mt * 32 + r * 16 + i + 8 * hv;
        const bf16_t bv = (bf16_t)(acc[r][g][i] + bias[col]);
        const int sel = col >> 9, c2 = col & 511;
        const int hh = c2 >> 6, cc = c2 & 63;
        const int bb = grow >> 11, n = grow & 2047;
        const size_t hb = (size_t)(bb * 8 + hh);
        if (sel == 0)      Qo[(hb * 2048 + n) * 64 + cc] = bv;
        else if (sel == 1) Ko[(hb * 2048 + n) * 64 + cc] = bv;
        else               Vto[(hb * 64 + cc) * 2048 + n] = bv;
      }
    }
  }
}

// ---------------------------------------------------------------------------
// Flash attention.  One block = 4 waves sharing one (b,h); each wave owns a
// 16-query tile.  K/V tiles (32 keys) are staged in LDS by async DMA, double
// buffered (ASYNCcnt); main loop unrolled x2 so the buffer index is constant.
// LDS rows padded (72 / 40 elems) for conflict-free 16-lane b128 ds_loads.
// ---------------------------------------------------------------------------
__global__ __launch_bounds__(128) void attn_kernel(const bf16_t* __restrict__ Q,
                                                   const bf16_t* __restrict__ K,
                                                   const bf16_t* __restrict__ Vt,
                                                   const unsigned char* __restrict__ mask,
                                                   bf16_t* __restrict__ O) {
  __shared__ __align__(16) bf16_t Kbuf[2][32 * 72];   // [key][dh], padded
  __shared__ __align__(16) bf16_t Vbuf[2][64 * 40];   // [dh][key], padded
  __shared__ __align__(16) bf16_t pstage[4][16 * 40]; // per-wave P C->A stage

  const int tid  = threadIdx.x;
  const int lane = tid & 31;
  const int wid  = tid >> 5;
  const int bh   = blockIdx.x >> 5;   // 64 (b,h) pairs
  const int qg   = blockIdx.x & 31;   // 32 query groups of 4 tiles
  const int tile = qg * 4 + wid;
  const int b    = bh >> 3;
  const int h    = bh & 7;
  const int hv   = lane >> 4, ln = lane & 15;

  const bf16_t* Qh = Q + ((size_t)bh * 2048 + tile * 16) * 64;
  const bf16_t* Kh = K + (size_t)bh * 2048 * 64;
  const bf16_t* Vh = Vt + (size_t)bh * 64 * 2048;
  bf16_t* ps = &pstage[wid][0];

  // stage one 32-key K tile + V tile: 4 async b128 per thread (2 K + 2 V)
  auto fill = [&](int buf, int k0) {
#pragma unroll
    for (int j = 0; j < 2; ++j) {
      const int idx = tid * 2 + j;              // 0..255
      const int kr = idx >> 3, kc = idx & 7;    // K: 32 rows x 128B
      async_cp16(&Kbuf[buf][kr * 72 + kc * 8],
                 Kh + (size_t)(k0 + kr) * 64 + kc * 8);
      const int vr = idx >> 2, vc = idx & 3;    // V: 64 rows x 64B
      async_cp16(&Vbuf[buf][vr * 40 + vc * 8],
                 Vh + (size_t)vr * 2048 + k0 + vc * 8);
    }
  };

  const v16bf a0 = load_tile(Qh, 64, lane);        // Q rows, dh 0..31
  const v16bf a1 = load_tile(Qh + 32, 64, lane);   // Q rows, dh 32..63

  bool qv[8];
#pragma unroll
  for (int i = 0; i < 8; ++i) {
    const int qrow = tile * 16 + i + 8 * hv;
    qv[i] = (qrow == 0) || (mask[(size_t)b * 2047 + qrow - 1] != 0);
  }

  float mrow[8], lrow[8];
  v8f o0 = {}, o1 = {}, o2 = {}, o3 = {};
#pragma unroll
  for (int i = 0; i < 8; ++i) { mrow[i] = -3.0e38f; lrow[i] = 0.f; }

  const float scale = 0.044194173824159216f;  // 512^-0.5

  // one 32-key step against LDS buffer `buf`
  auto step = [&](int buf, int kk) {
    if (kk + 32 >= 2048) { WAIT_ASYNC_0(); } else { WAIT_ASYNC_LE4(); }
    __syncthreads();   // all waves' fills of `buf` visible

    const bf16_t* kb = &Kbuf[buf][0];
    const bf16_t* vb = &Vbuf[buf][0];
    const v16bf b00 = load_tile(kb, 72, lane);             // keys 0-15, dh 0-31
    const v16bf b01 = load_tile(kb + 32, 72, lane);        // keys 0-15, dh 32-63
    const v16bf b10 = load_tile(kb + 16 * 72, 72, lane);
    const v16bf b11 = load_tile(kb + 16 * 72 + 32, 72, lane);

    v8f s0 = {}, s1 = {};
    s0 = WMMA_BF16(a0, b00, s0);
    s0 = WMMA_BF16(a1, b01, s0);
    s1 = WMMA_BF16(a0, b10, s1);
    s1 = WMMA_BF16(a1, b11, s1);

    const int c0 = kk + ln;
    const int c1 = c0 + 16;
    const bool kv0 = (c0 == 0) || (mask[(size_t)b * 2047 + c0 - 1] != 0);
    const bool kv1 = (mask[(size_t)b * 2047 + c1 - 1] != 0);

#pragma unroll
    for (int i = 0; i < 8; ++i) {
      float v0 = (kv0 && qv[i]) ? s0[i] * scale : -1.0e9f;
      float v1 = (kv1 && qv[i]) ? s1[i] * scale : -1.0e9f;
      float rmax = fmaxf(v0, v1);
#pragma unroll
      for (int m = 1; m < 16; m <<= 1) rmax = fmaxf(rmax, __shfl_xor(rmax, m, 32));
      const float mn   = fmaxf(mrow[i], rmax);
      const float p0   = __expf(v0 - mn);
      const float p1   = __expf(v1 - mn);
      const float corr = __expf(mrow[i] - mn);
      float rsum = p0 + p1;
#pragma unroll
      for (int m = 1; m < 16; m <<= 1) rsum += __shfl_xor(rsum, m, 32);
      lrow[i] = lrow[i] * corr + rsum;
      mrow[i] = mn;
      o0[i] *= corr; o1[i] *= corr; o2[i] *= corr; o3[i] *= corr;
      ps[(i + 8 * hv) * 40 + ln]      = (bf16_t)p0;
      ps[(i + 8 * hv) * 40 + 16 + ln] = (bf16_t)p1;
    }

    const v16bf pa  = load_tile(ps, 40, lane);
    const v16bf vb0 = load_tile(vb + 0 * 16 * 40, 40, lane);
    const v16bf vb1 = load_tile(vb + 1 * 16 * 40, 40, lane);
    const v16bf vb2 = load_tile(vb + 2 * 16 * 40, 40, lane);
    const v16bf vb3 = load_tile(vb + 3 * 16 * 40, 40, lane);
    o0 = WMMA_BF16(pa, vb0, o0);
    o1 = WMMA_BF16(pa, vb1, o1);
    o2 = WMMA_BF16(pa, vb2, o2);
    o3 = WMMA_BF16(pa, vb3, o3);

    __syncthreads();   // everyone done reading `buf` before it is refilled
    if (kk + 64 < 2048) fill(buf, kk + 64);
  };

  fill(0, 0);
  fill(1, 32);
  for (int k0 = 0; k0 < 2048; k0 += 64) {
    step(0, k0);
    step(1, k0 + 32);
  }

#pragma unroll
  for (int i = 0; i < 8; ++i) {
    const float rinv = 1.0f / lrow[i];
    const size_t row = (size_t)b * 2048 + tile * 16 + i + 8 * hv;
    bf16_t* dst = O + row * 512 + h * 64 + ln;
    dst[0]  = (bf16_t)(o0[i] * rinv);
    dst[16] = (bf16_t)(o1[i] * rinv);
    dst[32] = (bf16_t)(o2[i] * rinv);
    dst[48] = (bf16_t)(o3[i] * rinv);
  }
}

// ---------------------------------------------------------------------------
// Output projection: Ob[16384,512] @ Wout_t[512,512]^T + bout -> f32 out.
// 32x64 tile per wave, fully unrolled K loop.
// ---------------------------------------------------------------------------
__global__ __launch_bounds__(128) void oproj_kernel(const bf16_t* __restrict__ Ob,
                                                    const bf16_t* __restrict__ Wt,
                                                    const float* __restrict__ bias,
                                                    float* __restrict__ out) {
  const int lane = threadIdx.x & 31;
  const int wave = blockIdx.x * 4 + (threadIdx.x >> 5);
  const int mt = wave >> 3;            // 512 row tiles of 32
  const int nt = wave & 7;             // 8 col tiles of 64
  const int hv = lane >> 4, ln = lane & 15;
  const bf16_t* arow = Ob + (size_t)mt * 32 * 512;
  const bf16_t* wrow = Wt + (size_t)nt * 64 * 512;

  v8f acc[2][4] = {};
#pragma unroll
  for (int k0 = 0; k0 < 512; k0 += 32) {
    const v16bf a0t = load_tile(arow + k0, 512, lane);
    const v16bf a1t = load_tile(arow + (size_t)16 * 512 + k0, 512, lane);
#pragma unroll
    for (int g = 0; g < 4; ++g) {
      const v16bf w = load_tile(wrow + (size_t)g * 16 * 512 + k0, 512, lane);
      acc[0][g] = WMMA_BF16(a0t, w, acc[0][g]);
      acc[1][g] = WMMA_BF16(a1t, w, acc[1][g]);
    }
  }

#pragma unroll
  for (int r = 0; r < 2; ++r) {
#pragma unroll
    for (int g = 0; g < 4; ++g) {
#pragma unroll
      for (int i = 0; i < 8; ++i) {
        const int col  = nt * 64 + g * 16 + ln;
        const int grow = mt * 32 + r * 16 + i + 8 * hv;
        out[(size_t)grow * 512 + col] = acc[r][g][i] + bias[col];
      }
    }
  }
}

// ---------------------------------------------------------------------------
// Host launcher
// ---------------------------------------------------------------------------
extern "C" void kernel_launch(void* const* d_in, const int* in_sizes, int n_in,
                              void* d_out, int out_size, void* d_ws, size_t ws_size,
                              hipStream_t stream) {
  (void)in_sizes; (void)n_in; (void)out_size; (void)ws_size;
  const float* x            = (const float*)d_in[0];
  const unsigned char* mask = (const unsigned char*)d_in[1];  // jnp bool = 1 byte
  const float* Wqkv         = (const float*)d_in[2];
  const float* bqkv         = (const float*)d_in[3];
  const float* Wout         = (const float*)d_in[4];
  const float* bout         = (const float*)d_in[5];
  float* out                = (float*)d_out;

  char* ws = (char*)d_ws;
  const size_t SZ_XB    = (size_t)16384 * 512 * 2;
  const size_t SZ_WQKVT = (size_t)1536 * 512 * 2;
  const size_t SZ_WOUTT = (size_t)512 * 512 * 2;
  const size_t SZ_HEAD  = (size_t)8 * 8 * 2048 * 64 * 2;
  size_t off = 0;
  bf16_t* xb    = (bf16_t*)(ws + off); off += SZ_XB;
  bf16_t* Wqkvt = (bf16_t*)(ws + off); off += SZ_WQKVT;
  bf16_t* Woutt = (bf16_t*)(ws + off); off += SZ_WOUTT;
  bf16_t* Qb    = (bf16_t*)(ws + off); off += SZ_HEAD;
  bf16_t* Kb    = (bf16_t*)(ws + off); off += SZ_HEAD;
  bf16_t* Vtb   = (bf16_t*)(ws + off); off += SZ_HEAD;
  bf16_t* Obf   = (bf16_t*)(ws + off); off += SZ_XB;

  cvt_kernel<<<(16384 * 512 + 255) / 256, 256, 0, stream>>>(x, xb, 16384 * 512);
  cvtT_kernel<<<(512 * 1536 + 255) / 256, 256, 0, stream>>>(Wqkv, Wqkvt, 512, 1536);
  cvtT_kernel<<<(512 * 512 + 255) / 256, 256, 0, stream>>>(Wout, Woutt, 512, 512);

  // QKV: 512*24 wave tiles / 4 waves per block
  qkv_gemm_kernel<<<(512 * 24) / 4, 128, 0, stream>>>(xb, Wqkvt, bqkv, Qb, Kb, Vtb);
  // attention: 64 (b,h) * 32 query groups
  attn_kernel<<<2048, 128, 0, stream>>>(Qb, Kb, Vtb, mask, Obf);
  // out-proj: 512*8 wave tiles / 4 waves per block
  oproj_kernel<<<(512 * 8) / 4, 128, 0, stream>>>(Obf, Woutt, bout, out);
}
